// MatchingModel_GATv2SinkhornTopK_88098369175735
// MI455X (gfx1250) — compile-verified, matrix operand order checked by reading, not run on hardware
//
#include <hip/hip_runtime.h>
#include <hip/hip_bf16.h>

// ---------------------------------------------------------------------------
// MatchingModel GATv2 + Sinkhorn + soft-topk  — CDNA5 (gfx1250, wave32)
// ---------------------------------------------------------------------------
#define BGR   64          // graphs per batch
#define NPG   256         // nodes per graph
#define NN    (BGR*NPG)   // 16384 total nodes
#define EPG   4096        // edges per graph (random, no self loops)
#define ETOT  (BGR*EPG)   // 262144 edges per side
#define ELOOP NN          // appended self loops
#define EALL  (ETOT+ELOOP)
#define IN_DIM 7
#define HID    64
#define OUTD   32
#define NSQ    (NPG*NPG)      // 65536 scores per graph
#define NEG_SLOPE 0.2f
#define EPSN  1e-5f

typedef __attribute__((ext_vector_type(16))) _Float16 v16h;
typedef __attribute__((ext_vector_type(8)))  _Float16 v8h;
typedef __attribute__((ext_vector_type(8)))  float    v8f;

// ------------------------- wave32 reductions -------------------------------
__device__ __forceinline__ float wredSum(float v) {
#pragma unroll
  for (int o = 16; o > 0; o >>= 1) v += __shfl_xor(v, o, 32);
  return v;
}
__device__ __forceinline__ float wredMax(float v) {
#pragma unroll
  for (int o = 16; o > 0; o >>= 1) v = fmaxf(v, __shfl_xor(v, o, 32));
  return v;
}
__device__ __forceinline__ float wredMin(float v) {
#pragma unroll
  for (int o = 16; o > 0; o >>= 1) v = fminf(v, __shfl_xor(v, o, 32));
  return v;
}

__device__ __forceinline__ void atomicMaxF(float* addr, float val) {
  int* ia = (int*)addr;
  int old = __float_as_int(*addr);
  while (__int_as_float(old) < val) {
    int assumed = old;
    old = atomicCAS(ia, assumed, __float_as_int(val));
    if (old == assumed) break;
  }
}

// ------------------------- generic fill ------------------------------------
__global__ void k_fill(float* __restrict__ p, float v, int n) {
  int i = blockIdx.x * blockDim.x + threadIdx.x;
  if (i < n) p[i] = v;
}

// ---------------- node linear transforms: xl = xW_l + b_l, xr = xW_r + b_r --
__global__ void k_lin(const float* __restrict__ x,
                      const float* __restrict__ Wl, const float* __restrict__ bl,
                      const float* __restrict__ Wr, const float* __restrict__ br,
                      float* __restrict__ xl, float* __restrict__ xr,
                      int Din, int Dout, int total) {
  int idx = blockIdx.x * blockDim.x + threadIdx.x;
  if (idx >= total) return;
  int n = idx / Dout, d = idx - n * Dout;
  float al = bl[d], ar = br[d];
  const float* xr0 = x + (size_t)n * Din;
  for (int k = 0; k < Din; ++k) {
    float xv = xr0[k];
    al = fmaf(xv, Wl[k * Dout + d], al);
    ar = fmaf(xv, Wr[k * Dout + d], ar);
  }
  xl[idx] = al;
  xr[idx] = ar;
}

// -------- GATv2 edge score: e = att . leaky_relu(xl[src] + xr[dst]) --------
__global__ void k_edge_e(const int* __restrict__ src, const int* __restrict__ dst,
                         const float* __restrict__ xl, const float* __restrict__ xr,
                         const float* __restrict__ att,
                         float* __restrict__ e, float* __restrict__ emax, int D) {
  int j = blockIdx.x * blockDim.x + threadIdx.x;
  if (j >= EALL) return;
  int s, d;
  if (j < ETOT) { s = src[j]; d = dst[j]; } else { s = d = j - ETOT; }
  const float* pl = xl + (size_t)s * D;
  const float* pr = xr + (size_t)d * D;
  float acc = 0.f;
  for (int k = 0; k < D; ++k) {
    float m = pl[k] + pr[k];
    m = (m > 0.f) ? m : NEG_SLOPE * m;
    acc = fmaf(att[k], m, acc);
  }
  e[j] = acc;
  atomicMaxF(&emax[d], acc);
}

// -------- softmax numerator w = exp(e - emax[dst]); denom accumulation -----
__global__ void k_edge_w(const int* __restrict__ dst,
                         float* __restrict__ e, const float* __restrict__ emax,
                         float* __restrict__ denom) {
  int j = blockIdx.x * blockDim.x + threadIdx.x;
  if (j >= EALL) return;
  int d = (j < ETOT) ? dst[j] : (j - ETOT);
  float w = __expf(e[j] - emax[d]);
  e[j] = w;
  atomicAdd(&denom[d], w);
}

// -------- weighted scatter: out[dst] += (w/denom[dst]) * xl[src] -----------
__global__ void k_edge_agg(const int* __restrict__ src, const int* __restrict__ dst,
                           const float* __restrict__ w, const float* __restrict__ denom,
                           const float* __restrict__ xl, float* __restrict__ out, int D) {
  int j = blockIdx.x * blockDim.x + threadIdx.x;
  if (j >= EALL) return;
  int s, d;
  if (j < ETOT) { s = src[j]; d = dst[j]; } else { s = d = j - ETOT; }
  float alpha = w[j] / denom[d];
  const float* pl = xl + (size_t)s * D;
  float* po = out + (size_t)d * D;
  for (int k = 0; k < D; ++k) atomicAdd(&po[k], alpha * pl[k]);
}

// -------- bias (+ relu) in place (layer-1 epilogue) ------------------------
__global__ void k_bias_act(float* __restrict__ h, const float* __restrict__ bias,
                           int D, int total, int relu) {
  int idx = blockIdx.x * blockDim.x + threadIdx.x;
  if (idx >= total) return;
  float v = h[idx] + bias[idx % D];
  if (relu) v = fmaxf(v, 0.f);
  h[idx] = v;
}

// -------- layer-2 epilogue: bias add + f16 materialization for WMMA --------
__global__ void k_bias_f16(const float* __restrict__ h, const float* __restrict__ bias,
                           _Float16* __restrict__ out16, int D, int total) {
  int idx = blockIdx.x * blockDim.x + threadIdx.x;
  if (idx >= total) return;
  out16[idx] = (_Float16)(h[idx] + bias[idx % D]);
}

// ---------------------------------------------------------------------------
// sim[b] = hA[b] (256x32) @ hB[b]^T  via V_WMMA_F32_16X16X32_F16.
// Operands are pre-converted f16; each lane's K pattern (k = half*8+e and
// 16+half*8+e, per the ISA 16-bit A layout) is two contiguous 16B chunks,
// so each operand is two b128-class loads — no conversion VALU in the loop.
// Each wave computes a 16x64 strip: A loaded once, 4 WMMAs (4 N-tiles).
// Fused per-graph sum / sum^2 reduction for InstanceNorm.
// grid = B*16*4 waves / 8 waves per block = 512 blocks x 256 threads.
// ---------------------------------------------------------------------------
__global__ void k_sim_wmma(const _Float16* __restrict__ hA, const _Float16* __restrict__ hB,
                           float* __restrict__ sim,
                           float* __restrict__ sums, float* __restrict__ sumsq) {
  int wave = blockIdx.x * 8 + (threadIdx.x >> 5);
  int lane = threadIdx.x & 31;
  int b    = wave >> 6;          // graph
  int tm   = (wave >> 2) & 15;   // row tile
  int tq   = wave & 3;           // N-quadrant (4 tiles each)
  int half = lane >> 4;          // K-half select (ISA 16-bit A layout)
  int sub  = lane & 15;          // M (resp. N) within tile

  const _Float16* arow = hA + (size_t)(b * NPG + tm * 16 + sub) * OUTD;
  v8h alo = *(const v8h*)(arow + half * 8);        // k = half*8 .. +7
  v8h ahi = *(const v8h*)(arow + 16 + half * 8);   // k = 16+half*8 .. +7
  v16h a;
#pragma unroll
  for (int e = 0; e < 8; ++e) { a[e] = alo[e]; a[e + 8] = ahi[e]; }

  float s = 0.f, s2 = 0.f;
#pragma unroll
  for (int j = 0; j < 4; ++j) {
    int tn = tq * 4 + j;
    const _Float16* brow = hB + (size_t)(b * NPG + tn * 16 + sub) * OUTD;
    v8h blo = *(const v8h*)(brow + half * 8);
    v8h bhi = *(const v8h*)(brow + 16 + half * 8);
    v16h bm;
#pragma unroll
    for (int e = 0; e < 8; ++e) { bm[e] = blo[e]; bm[e + 8] = bhi[e]; }

    v8f c = {};
    c = __builtin_amdgcn_wmma_f32_16x16x32_f16(false, a, false, bm,
                                               (short)0, c, false, false);

    float* outp = sim + (size_t)b * NSQ + (size_t)(tm * 16) * NPG + tn * 16;
#pragma unroll
    for (int r = 0; r < 8; ++r) {
      int m = r + half * 8;            // ISA C/D layout: lanes 16-31 -> M+8
      float v = c[r];
      outp[m * NPG + sub] = v;
      s += v;
      s2 += v * v;
    }
  }
  s  = wredSum(s);
  s2 = wredSum(s2);
  if (lane == 0) {
    atomicAdd(&sums[b], s);
    atomicAdd(&sumsq[b], s2);
  }
}

// ---------------------------------------------------------------------------
// Sinkhorn log-domain row normalization (axis=2). First pass fuses
// InstanceNorm: log_s = (gamma*(x-mean)*rsqrt(var+eps)+beta)/tau, tau=1.
// One wave per row, 8 rows per 256-thread block.
// ---------------------------------------------------------------------------
__global__ void k_rownorm(float* __restrict__ s,
                          const float* __restrict__ sums, const float* __restrict__ sumsq,
                          const float* __restrict__ gammaP, const float* __restrict__ betaP,
                          int applyIN) {
  int wid = threadIdx.x >> 5, lane = threadIdx.x & 31;
  int row = blockIdx.x * 8 + wid;      // 0..16383
  int b   = row >> 8;
  float* p = s + (size_t)row * NPG;

  float scale = 1.f, shift = 0.f;
  if (applyIN) {
    float mean = sums[b] * (1.f / (float)NSQ);
    float var  = sumsq[b] * (1.f / (float)NSQ) - mean * mean;
    float inv  = rsqrtf(var + EPSN);
    float gm = gammaP[0], bt = betaP[0];
    scale = gm * inv;
    shift = bt - gm * inv * mean;
  }

  float v[8];
  float mx = -3.4e38f;
#pragma unroll
  for (int i = 0; i < 8; ++i) {
    v[i] = fmaf(p[lane + 32 * i], scale, shift);
    mx = fmaxf(mx, v[i]);
  }
  mx = wredMax(mx);
  float sm = 0.f;
#pragma unroll
  for (int i = 0; i < 8; ++i) sm += __expf(v[i] - mx);
  sm = wredSum(sm);
  float lse = mx + __logf(sm);
#pragma unroll
  for (int i = 0; i < 8; ++i) p[lane + 32 * i] = v[i] - lse;
}

// Column normalization (axis=1). Last pass writes exp() (the S matrix).
__global__ void k_colnorm(float* __restrict__ s, int writeExp) {
  int wid = threadIdx.x >> 5, lane = threadIdx.x & 31;
  int col = blockIdx.x * 8 + wid;      // 0..16383
  int b = col >> 8, c = col & 255;
  float* base = s + (size_t)b * NSQ + c;

  float v[8];
  float mx = -3.4e38f;
#pragma unroll
  for (int i = 0; i < 8; ++i) {
    v[i] = base[(lane + 32 * i) * NPG];
    mx = fmaxf(mx, v[i]);
  }
  mx = wredMax(mx);
  float sm = 0.f;
#pragma unroll
  for (int i = 0; i < 8; ++i) sm += __expf(v[i] - mx);
  sm = wredSum(sm);
  float lse = mx + __logf(sm);
#pragma unroll
  for (int i = 0; i < 8; ++i) {
    float r = v[i] - lse;
    base[(lane + 32 * i) * NPG] = writeExp ? __expf(r) : r;
  }
}

// ---------------------------------------------------------------------------
// soft-topk: OT between the 65536 scores of one graph and 2 anchors
// (min, max).  One 256-thread block per graph, in-place on d_out.
// ---------------------------------------------------------------------------
__global__ void k_topk(float* __restrict__ sAll) {
  __shared__ float red0[8], red1[8], shv[4];  // a0, a1, g0, g1
  int t = threadIdx.x, lane = t & 31, wid = t >> 5;
  float* x = sAll + (size_t)blockIdx.x * NSQ;

  // anchors = min / max over the graph's scores
  float mn = 3.4e38f, mx = -3.4e38f;
  for (int i = 0; i < 256; ++i) {
    float v = x[i * 256 + t];
    mn = fminf(mn, v);
    mx = fmaxf(mx, v);
  }
  mn = wredMin(mn);
  mx = wredMax(mx);
  if (lane == 0) { red0[wid] = mn; red1[wid] = mx; }
  __syncthreads();
  if (t == 0) {
    float a = red0[0], bm = red1[0];
    for (int i = 1; i < 8; ++i) { a = fminf(a, red0[i]); bm = fmaxf(bm, red1[i]); }
    shv[0] = a; shv[1] = bm; shv[2] = 0.f; shv[3] = 0.f;
  }
  __syncthreads();
  float a0 = shv[0], a1 = shv[1];

  const float log_mu  = -__logf((float)NSQ);
  const float log_nu0 = __logf((float)(NSQ - NPG) / (float)NSQ);
  const float log_nu1 = __logf((float)NPG / (float)NSQ);

  float g0p = 0.f, g1p = 0.f;  // g at the start of the final iteration (g_9)
  for (int it = 0; it < 10; ++it) {
    float g0 = shv[2], g1 = shv[3];
    g0p = g0; g1p = g1;
    float s0 = 0.f, s1 = 0.f;
    for (int i = 0; i < 256; ++i) {
      float v  = x[i * 256 + t];
      float l0 = -fabsf(v - a0), l1 = -fabsf(v - a1);
      float t0 = l0 + g0, t1 = l1 + g1;
      float m  = fmaxf(t0, t1);
      float f  = log_mu - (m + __logf(__expf(t0 - m) + __expf(t1 - m)));
      s0 += __expf(l0 + f);
      s1 += __expf(l1 + f);
    }
    s0 = wredSum(s0);
    s1 = wredSum(s1);
    if (lane == 0) { red0[wid] = s0; red1[wid] = s1; }
    __syncthreads();
    if (t == 0) {
      float S0 = 0.f, S1 = 0.f;
      for (int i = 0; i < 8; ++i) { S0 += red0[i]; S1 += red1[i]; }
      shv[2] = log_nu0 - __logf(S0);
      shv[3] = log_nu1 - __logf(S1);
    }
    __syncthreads();
  }
  float g1f = shv[3];  // final g (g_10); f below uses g_9 like the reference
  for (int i = 0; i < 256; ++i) {
    float v  = x[i * 256 + t];
    float l0 = -fabsf(v - a0), l1 = -fabsf(v - a1);
    float t0 = l0 + g0p, t1 = l1 + g1p;
    float m  = fmaxf(t0, t1);
    float f  = log_mu - (m + __logf(__expf(t0 - m) + __expf(t1 - m)));
    x[i * 256 + t] = __expf(l1 + f + g1f) * (float)NSQ;
  }
}

// ---------------------------------------------------------------------------
extern "C" void kernel_launch(void* const* d_in, const int* in_sizes, int n_in,
                              void* d_out, int out_size, void* d_ws, size_t ws_size,
                              hipStream_t stream) {
  (void)in_sizes; (void)n_in; (void)out_size; (void)ws_size;

  const float* x1    = (const float*)d_in[0];
  const int*   ei1   = (const int*)  d_in[1];
  const float* x2    = (const float*)d_in[2];
  const int*   ei2   = (const int*)  d_in[3];
  const float* Wl1   = (const float*)d_in[4];
  const float* bl1   = (const float*)d_in[5];
  const float* Wr1   = (const float*)d_in[6];
  const float* br1   = (const float*)d_in[7];
  const float* att1  = (const float*)d_in[8];
  const float* bias1 = (const float*)d_in[9];
  const float* Wl2   = (const float*)d_in[10];
  const float* bl2   = (const float*)d_in[11];
  const float* Wr2   = (const float*)d_in[12];
  const float* br2   = (const float*)d_in[13];
  const float* att2  = (const float*)d_in[14];
  const float* bias2 = (const float*)d_in[15];
  const float* ingam = (const float*)d_in[16];
  const float* inbet = (const float*)d_in[17];

  float* out = (float*)d_out;

  // Scratch staged inside d_out (fully overwritten by k_sim_wmma later):
  float* XL = out;                 // [NN,64]
  float* XR = out + NN * HID;      // [NN,64]
  float* H1 = out + 2 * NN * HID;  // [NN,64]

  // Real workspace (~5 MB)
  float*     agg   = (float*)d_ws;                 // [NN,32] f32 aggregation (shared)
  _Float16*  hA16  = (_Float16*)(agg + NN * OUTD); // [NN,32] f16
  _Float16*  hB16  = hA16 + NN * OUTD;             // [NN,32] f16
  float*     eBuf  = (float*)(hB16 + NN * OUTD);   // [EALL]
  float*     emax  = eBuf + EALL;                  // [NN]
  float*     denom = emax + NN;                    // [NN]
  float*     sums  = denom + NN;                   // [B]
  float*     sumsq = sums + BGR;                   // [B]

  const int T = 256;
  const int gLin1 = (NN * HID + T - 1) / T;
  const int gLin2 = (NN * OUTD + T - 1) / T;
  const int gEdge = (EALL + T - 1) / T;
  const int gNode = (NN + T - 1) / T;

  for (int side = 0; side < 2; ++side) {
    const float* x   = side ? x2 : x1;
    const int*   src = side ? ei2 : ei1;
    const int*   dst = src + ETOT;
    _Float16* hFin16 = side ? hB16 : hA16;

    // ---- layer 1 (7 -> 64) ----
    k_lin<<<gLin1, T, 0, stream>>>(x, Wl1, bl1, Wr1, br1, XL, XR, IN_DIM, HID, NN * HID);
    k_fill<<<gLin1, T, 0, stream>>>(H1, 0.f, NN * HID);
    k_fill<<<gNode, T, 0, stream>>>(emax, -3.0e38f, NN);
    k_fill<<<gNode, T, 0, stream>>>(denom, 0.f, NN);
    k_edge_e  <<<gEdge, T, 0, stream>>>(src, dst, XL, XR, att1, eBuf, emax, HID);
    k_edge_w  <<<gEdge, T, 0, stream>>>(dst, eBuf, emax, denom);
    k_edge_agg<<<gEdge, T, 0, stream>>>(src, dst, eBuf, denom, XL, H1, HID);
    k_bias_act<<<gLin1, T, 0, stream>>>(H1, bias1, HID, NN * HID, 1);

    // ---- layer 2 (64 -> 32) ----
    k_lin<<<gLin2, T, 0, stream>>>(H1, Wl2, bl2, Wr2, br2, XL, XR, HID, OUTD, NN * OUTD);
    k_fill<<<gLin2, T, 0, stream>>>(agg, 0.f, NN * OUTD);
    k_fill<<<gNode, T, 0, stream>>>(emax, -3.0e38f, NN);
    k_fill<<<gNode, T, 0, stream>>>(denom, 0.f, NN);
    k_edge_e  <<<gEdge, T, 0, stream>>>(src, dst, XL, XR, att2, eBuf, emax, OUTD);
    k_edge_w  <<<gEdge, T, 0, stream>>>(dst, eBuf, emax, denom);
    k_edge_agg<<<gEdge, T, 0, stream>>>(src, dst, eBuf, denom, XL, agg, OUTD);
    k_bias_f16<<<gLin2, T, 0, stream>>>(agg, bias2, hFin16, OUTD, NN * OUTD);
  }

  // ---- batched similarity GEMM via WMMA + fused InstanceNorm stats ----
  k_fill<<<1, 128, 0, stream>>>(sums, 0.f, BGR);
  k_fill<<<1, 128, 0, stream>>>(sumsq, 0.f, BGR);
  // B*16*4 = 4096 waves, 8 waves per 256-thread block -> 512 blocks
  k_sim_wmma<<<(BGR * 16 * 4) / 8, T, 0, stream>>>(hA16, hB16, out, sums, sumsq);

  // ---- 10 Sinkhorn iterations: row (axis=2) on even, col (axis=1) on odd --
  const int gRC = (BGR * NPG) / 8;  // 2048 blocks, 8 rows/cols per block
  for (int it = 0; it < 10; ++it) {
    if ((it & 1) == 0)
      k_rownorm<<<gRC, T, 0, stream>>>(out, sums, sumsq, ingam, inbet, it == 0 ? 1 : 0);
    else
      k_colnorm<<<gRC, T, 0, stream>>>(out, it == 9 ? 1 : 0);
  }

  // ---- soft-topk (one block per graph, in place) ----
  k_topk<<<BGR, T, 0, stream>>>(out);
}
